// GAT_52999896432942
// MI455X (gfx1250) — matrix operand order repaired and checked by reading, not compiled, not run on hardware
//
#include <hip/hip_runtime.h>

typedef __attribute__((ext_vector_type(16))) _Float16 v16h;
typedef __attribute__((ext_vector_type(8)))  _Float16 v8h;
typedef __attribute__((ext_vector_type(8)))  float    v8f;

#define FIN   128
#define NH1   8
#define ND1   32
#define HD1   256   // NH1*ND1
#define NCLS  40
#define NCLSP 48    // padded to 3 WMMA column tiles

// ---------------- prep kernels ----------------

__global__ void k_f32_to_f16(const float* __restrict__ in, _Float16* __restrict__ out, int n) {
  int i = blockIdx.x * blockDim.x + threadIdx.x;
  if (i < n) out[i] = (_Float16)in[i];
}

__global__ void k_relu_f32_to_f16(const float* __restrict__ in, _Float16* __restrict__ out, int n) {
  int i = blockIdx.x * blockDim.x + threadIdx.x;
  if (i < n) out[i] = (_Float16)fmaxf(in[i], 0.f);
}

// out[Cout x R] = transpose(in[R x Cin]) with zero-padded extra columns
__global__ void k_transpose_pad_f16(const float* __restrict__ in, _Float16* __restrict__ out,
                                    int R, int Cin, int Cout) {
  int i = blockIdx.x * blockDim.x + threadIdx.x;
  if (i >= R * Cout) return;
  int c = i / R, r = i - c * R;
  out[(size_t)c * R + r] = (c < Cin) ? (_Float16)in[(size_t)r * Cin + c] : (_Float16)0.f;
}

// ---------------- WMMA GEMM: C[M x ldc] = A[M x K] * B[K x ldc], BT = B^T [ldc x K] ----
// one wave per 16x(16*NT) tile group: A loaded once per k-step, NT WMMAs issued on it.
// grid = (ldc/(16*NT), M/16), block = 32

template <int NT>
__global__ void k_wmma_gemm_nt(const _Float16* __restrict__ A, const _Float16* __restrict__ BT,
                               float* __restrict__ C, int K, int ldc) {
  const int ng   = blockIdx.x;     // column-tile group
  const int mt   = blockIdx.y;     // row tile
  const int lane = threadIdx.x & 31;
  const int hsel = lane >> 4;      // 0: lanes 0-15, 1: lanes 16-31
  const int l16  = lane & 15;

  // A layout (16-bit 16x32): lane holds M=l16; K halves split by hsel (0..7/16..23 vs 8..15/24..31)
  const _Float16* arow  = A  + (size_t)(mt * 16 + l16) * K + hsel * 8;
  // B layout (16-bit 32x16): lane holds N=l16; K halves split by hsel (0..7/16..23 vs 8..15/24..31) (BT row-major)
  const _Float16* brow0 = BT + (size_t)(ng * NT * 16 + l16) * K + hsel * 8;

  v8f acc[NT];
#pragma unroll
  for (int t = 0; t < NT; ++t) acc[t] = (v8f){};

  for (int k0 = 0; k0 < K; k0 += 32) {
    v8h alo = *(const v8h*)(arow + k0);
    v8h ahi = *(const v8h*)(arow + k0 + 16);
    v16h a = __builtin_shufflevector(alo, ahi, 0,1,2,3,4,5,6,7,8,9,10,11,12,13,14,15);
#pragma unroll
    for (int t = 0; t < NT; ++t) {
      const _Float16* brow = brow0 + (size_t)t * 16 * K;
      v8h blo = *(const v8h*)(brow + k0);
      v8h bhi = *(const v8h*)(brow + k0 + 16);
      v16h b = __builtin_shufflevector(blo, bhi, 0,1,2,3,4,5,6,7,8,9,10,11,12,13,14,15);
      acc[t] = __builtin_amdgcn_wmma_f32_16x16x32_f16(false, a, false, b, (short)0,
                                                      acc[t], false, false);
    }
  }
  // C/D layout: VGPR r -> row r + 8*hsel, col l16
#pragma unroll
  for (int t = 0; t < NT; ++t) {
    float* cbase = C + (size_t)(mt * 16 + hsel * 8) * ldc + (ng * NT + t) * 16 + l16;
#pragma unroll
    for (int r = 0; r < 8; ++r) cbase[(size_t)r * ldc] = acc[t][r];
  }
}

// ---------------- GAT attention kernels ----------------

// el/er per (node, head): dot(ft[n,h,:], attn)
__global__ void k_node_scores(const float* __restrict__ ft, const float* __restrict__ al,
                              const float* __restrict__ ar, float* __restrict__ el,
                              float* __restrict__ er, int nH, int H, int D, int ld) {
  int i = blockIdx.x * blockDim.x + threadIdx.x;
  if (i >= nH) return;
  int n = i / H, h = i - n * H;
  const float* f = ft + (size_t)n * ld + h * D;
  float sl = 0.f, sr = 0.f;
  for (int d = 0; d < D; ++d) { float v = f[d]; sl += v * al[h * D + d]; sr += v * ar[h * D + d]; }
  el[i] = sl; er[i] = sr;
}

__device__ __forceinline__ unsigned f32_key(float f) {
  unsigned b = __float_as_uint(f);
  return (b & 0x80000000u) ? ~b : (b | 0x80000000u);
}
__device__ __forceinline__ float key_f32(unsigned k) {
  unsigned b = (k & 0x80000000u) ? (k & 0x7FFFFFFFu) : ~k;
  return __uint_as_float(b);
}

// pass 1: e = leaky_relu(el[src]+er[dst]); segment-max via uint-encoded atomicMax
__global__ void k_edge_score(const float* __restrict__ el, const float* __restrict__ er,
                             const int* __restrict__ src, const int* __restrict__ dst,
                             float* __restrict__ e_out, unsigned* __restrict__ mkey,
                             int EH, int H) {
  int i = blockIdx.x * blockDim.x + threadIdx.x;
  if (i >= EH) return;
  int e = i / H, h = i - e * H;
  int dn = dst[e];
  float v = el[(size_t)src[e] * H + h] + er[(size_t)dn * H + h];
  v = v > 0.f ? v : 0.2f * v;
  e_out[i] = v;
  atomicMax(&mkey[(size_t)dn * H + h], f32_key(v));
}

// pass 2: ex = exp(e - m[dst]); segment-sum
__global__ void k_edge_exp(const int* __restrict__ dst, const unsigned* __restrict__ mkey,
                           float* __restrict__ e_io, float* __restrict__ s, int EH, int H) {
  int i = blockIdx.x * blockDim.x + threadIdx.x;
  if (i >= EH) return;
  int e = i / H, h = i - e * H;
  int dn = dst[e];
  float m = key_f32(mkey[(size_t)dn * H + h]);
  if (!__builtin_isfinite(m)) m = 0.f;           // mirrors reference guard
  float ex = __expf(e_io[i] - m);
  e_io[i] = ex;
  atomicAdd(&s[(size_t)dn * H + h], ex);
}

// pass 3: out[dst] += ft[src] * alpha ; one thread per (edge, 4 consecutive features)
__global__ void k_edge_agg(const float* __restrict__ ft, const float* __restrict__ ex,
                           const float* __restrict__ s, const int* __restrict__ src,
                           const int* __restrict__ dst, float* __restrict__ out,
                           long long total4, int H, int D, int ldin, int ldout) {
  long long i = (long long)blockIdx.x * blockDim.x + threadIdx.x;
  if (i >= total4) return;
  long long f = i * 4;
  int HD = H * D;
  int e = (int)(f / HD);
  int r = (int)(f - (long long)e * HD);
  int h = r / D;
  int sn = src[e], dn = dst[e];
  float alpha = ex[(size_t)e * H + h] / fmaxf(s[(size_t)dn * H + h], 1e-16f);
  const float* fp = ft + (size_t)sn * ldin + r;
  float* op = out + (size_t)dn * ldout + r;
#pragma unroll
  for (int j = 0; j < 4; ++j) atomicAdd(op + j, fp[j] * alpha);
}

// ---------------- host ----------------

extern "C" void kernel_launch(void* const* d_in, const int* in_sizes, int n_in,
                              void* d_out, int out_size, void* d_ws, size_t ws_size,
                              hipStream_t stream) {
  (void)n_in; (void)out_size; (void)ws_size;
  const float* x   = (const float*)d_in[0];
  const float* W1  = (const float*)d_in[1];
  const float* al1 = (const float*)d_in[2];
  const float* ar1 = (const float*)d_in[3];
  const float* W2  = (const float*)d_in[4];
  const float* al2 = (const float*)d_in[5];
  const float* ar2 = (const float*)d_in[6];
  const int*   src = (const int*)d_in[7];
  const int*   dst = (const int*)d_in[8];
  float* out = (float*)d_out;

  const int E = in_sizes[7];
  const int N = in_sizes[0] / FIN;   // 20000 (multiple of 16)

  char* ws = (char*)d_ws;
  size_t off = 0;
  auto alloc = [&](size_t bytes) -> void* {
    void* p = ws + off; off = (off + bytes + 255) & ~(size_t)255; return p;
  };
  _Float16* Xh   = (_Float16*)alloc((size_t)N * FIN * 2);
  _Float16* W1T  = (_Float16*)alloc((size_t)HD1 * FIN * 2);
  float*    ft1  = (float*)   alloc((size_t)N * HD1 * 4);
  float*    el1  = (float*)   alloc((size_t)N * NH1 * 4);
  float*    er1  = (float*)   alloc((size_t)N * NH1 * 4);
  float*    e1   = (float*)   alloc((size_t)E * NH1 * 4);
  unsigned* m1   = (unsigned*)alloc((size_t)N * NH1 * 4);
  float*    s1   = (float*)   alloc((size_t)N * NH1 * 4);
  float*    out1 = (float*)   alloc((size_t)N * HD1 * 4);
  _Float16* H1h  = (_Float16*)alloc((size_t)N * HD1 * 2);
  _Float16* W2T  = (_Float16*)alloc((size_t)NCLSP * HD1 * 2);
  float*    ft2  = (float*)   alloc((size_t)N * NCLSP * 4);
  float*    el2  = (float*)   alloc((size_t)N * 4);
  float*    er2  = (float*)   alloc((size_t)N * 4);
  float*    e2   = (float*)   alloc((size_t)E * 4);
  unsigned* m2   = (unsigned*)alloc((size_t)N * 4);
  float*    s2   = (float*)   alloc((size_t)N * 4);

  const int B = 256;
  auto G = [](long long n, int b) { return (unsigned)((n + b - 1) / b); };

  hipMemsetAsync(m1,   0, (size_t)N * NH1 * 4, stream);
  hipMemsetAsync(s1,   0, (size_t)N * NH1 * 4, stream);
  hipMemsetAsync(out1, 0, (size_t)N * HD1 * 4, stream);
  hipMemsetAsync(m2,   0, (size_t)N * 4, stream);
  hipMemsetAsync(s2,   0, (size_t)N * 4, stream);
  hipMemsetAsync(out,  0, (size_t)N * NCLS * 4, stream);

  // ---- layer 1 ----
  k_f32_to_f16<<<G((long long)N * FIN, B), B, 0, stream>>>(x, Xh, N * FIN);
  k_transpose_pad_f16<<<G((long long)FIN * HD1, B), B, 0, stream>>>(W1, W1T, FIN, HD1, HD1);

  // 256 columns = 4 groups of 4 tiles
  k_wmma_gemm_nt<4><<<dim3(HD1 / 64, N / 16), 32, 0, stream>>>(Xh, W1T, ft1, FIN, HD1);

  k_node_scores<<<G((long long)N * NH1, B), B, 0, stream>>>(ft1, al1, ar1, el1, er1,
                                                            N * NH1, NH1, ND1, HD1);
  k_edge_score<<<G((long long)E * NH1, B), B, 0, stream>>>(el1, er1, src, dst, e1, m1,
                                                           E * NH1, NH1);
  k_edge_exp<<<G((long long)E * NH1, B), B, 0, stream>>>(dst, m1, e1, s1, E * NH1, NH1);
  k_edge_agg<<<G((long long)E * (HD1 / 4), B), B, 0, stream>>>(ft1, e1, s1, src, dst, out1,
                                                               (long long)E * (HD1 / 4),
                                                               NH1, ND1, HD1, HD1);

  // ---- layer 2 ----
  k_relu_f32_to_f16<<<G((long long)N * HD1, B), B, 0, stream>>>(out1, H1h, N * HD1);
  k_transpose_pad_f16<<<G((long long)HD1 * NCLSP, B), B, 0, stream>>>(W2, W2T, HD1, NCLS, NCLSP);

  // 48 columns = 1 group of 3 tiles
  k_wmma_gemm_nt<3><<<dim3(NCLSP / 48, N / 16), 32, 0, stream>>>(H1h, W2T, ft2, HD1, NCLSP);

  k_node_scores<<<G((long long)N, B), B, 0, stream>>>(ft2, al2, ar2, el2, er2,
                                                      N, 1, NCLS, NCLSP);
  k_edge_score<<<G((long long)E, B), B, 0, stream>>>(el2, er2, src, dst, e2, m2, E, 1);
  k_edge_exp<<<G((long long)E, B), B, 0, stream>>>(dst, m2, e2, s2, E, 1);
  k_edge_agg<<<G((long long)E * (NCLS / 4), B), B, 0, stream>>>(ft2, e2, s2, src, dst, out,
                                                                (long long)E * (NCLS / 4),
                                                                1, NCLS, NCLSP, NCLS);
}
